// LSTMModel_20005957665541
// MI455X (gfx1250) — matrix operand order repaired and checked by reading, not compiled, not run on hardware
//
#include <hip/hip_runtime.h>

// ---------------------------------------------------------------------------
// LSTM LM on gfx1250 (MI455X): bf16 WMMA everywhere, f32 accumulation.
//   VOCAB=32000, B=32, T=128, H=E=128, 4H=512
// Pipeline:
//   1) transpose+convert weights W0,U0,W1,U1,Wd -> N-major bf16 (scratch)
//   2) embedding gather -> x bf16 [B*T,128]
//   3) Z0 = x@W0 + b0  (parallel WMMA GEMM, stored [T][32][512] f32)
//   4) persistent single-workgroup scan over T with double-buffered
//      GLOBAL_LOAD_ASYNC_TO_LDS prefetch of the next Z0 slab:
//        z0 = Z0[t] + h0@U0 ; gates ; (c0,h0)
//        z1 = h0@W1 + h1@U1 + b1 ; gates ; (c1,h1) -> y bf16 [B*T,128]
//   5) logits = y@Wd + bd (WMMA GEMM; 524MB f32 output streamed with
//      non-temporal stores so L2 stays reserved for the reused operands)
// ---------------------------------------------------------------------------

typedef __attribute__((ext_vector_type(16))) __bf16 v16bf;
typedef __attribute__((ext_vector_type(8)))  float  v8f;

#define Bsz 32
#define Tsz 128
#define Hsz 128
#define Gsz 512      // 4*H
#define Vsz 32000

__device__ __forceinline__ unsigned short f32_to_bf16(float f) {
    unsigned int u = __float_as_uint(f);
    unsigned int r = u + 0x7FFFu + ((u >> 16) & 1u);   // round-to-nearest-even
    return (unsigned short)(r >> 16);
}

// A-fragment (16x32 bf16, row-major source, contiguous k order, same as B):
// lane holds row (lane&15); 16 consecutive K starting at (lane>>4)*16
__device__ __forceinline__ v16bf load_a_frag(const unsigned short* p) {
    v16bf r;
    ((uint4*)&r)[0] = *(const uint4*)(p);
    ((uint4*)&r)[1] = *(const uint4*)(p + 8);
    return r;
}
// B-fragment (32x16 bf16 from N-major (transposed) source):
// lane holds col (lane&15); 16 consecutive K starting at (lane>>4)*16
__device__ __forceinline__ v16bf load_b_frag(const unsigned short* p) {
    v16bf r;
    ((uint4*)&r)[0] = *(const uint4*)(p);
    ((uint4*)&r)[1] = *(const uint4*)(p + 8);
    return r;
}

__device__ __forceinline__ v8f wmma_bf16(v16bf a, v16bf b, v8f c) {
    return __builtin_amdgcn_wmma_f32_16x16x32_bf16(false, a, false, b, (short)0, c,
                                                   false, false);
}

__device__ __forceinline__ float sigm(float x) { return 1.0f / (1.0f + __expf(-x)); }

// ---------------- weight transpose + f32->bf16 convert ----------------
// in[K][N] f32 -> out[N][K] bf16
__global__ void k_transpose_bf16(const float* __restrict__ in,
                                 unsigned short* __restrict__ out, int K, int N) {
    int i = blockIdx.x * blockDim.x + threadIdx.x;
    int total = K * N;
    if (i >= total) return;
    int n = i / K;
    int k = i - n * K;
    out[i] = f32_to_bf16(in[k * N + n]);
}

// ---------------- embedding gather -> bf16 ----------------
__global__ void k_embed(const int* __restrict__ tok, const float* __restrict__ emb,
                        unsigned short* __restrict__ x, int total) {
    int i = blockIdx.x * blockDim.x + threadIdx.x;
    if (i >= total) return;
    int bt = i >> 7;
    int e  = i & 127;
    x[i] = f32_to_bf16(emb[(size_t)tok[bt] * Hsz + e]);
}

// ---------------- Z0 = x @ W0 + b0 ----------------
// X: [4096,128] bf16 row-major (row = b*T + t); W0t: [512,128] bf16 N-major
// Z0 stored [T][32][512] f32 so the scan reads it contiguously per step.
__global__ void k_gemm_z0(const unsigned short* __restrict__ X,
                          const unsigned short* __restrict__ W0t,
                          const float* __restrict__ b0, float* __restrict__ Z0) {
    int lane = threadIdx.x & 31;
    int wave = threadIdx.x >> 5;
    int tile = blockIdx.x * 4 + wave;         // 256*32 = 8192 tiles
    int tm = tile >> 5;                        // 0..255
    int tn = tile & 31;                        // 0..31
    int lo = lane & 15, hi = lane >> 4;
    int n = tn * 16 + lo;

    float bias = b0[n];
    v8f acc;
#pragma unroll
    for (int r = 0; r < 8; ++r) acc[r] = bias;

    const unsigned short* ab = X   + (size_t)(tm * 16 + lo) * 128 + hi * 16;
    const unsigned short* bb = W0t + (size_t)(tn * 16 + lo) * 128 + hi * 16;
#pragma unroll
    for (int kk = 0; kk < 4; ++kk) {
        v16bf a = load_a_frag(ab + kk * 32);
        v16bf b = load_b_frag(bb + kk * 32);
        acc = wmma_bf16(a, b, acc);
    }
#pragma unroll
    for (int r = 0; r < 8; ++r) {
        int M  = tm * 16 + r + hi * 8;         // bt row
        int bi = M >> 7, ti = M & 127;
        Z0[((size_t)ti * Bsz + bi) * Gsz + n] = acc[r];
    }
}

// ---------------- persistent LSTM scan (1 workgroup, 16 waves) ----------------
// LDS: z_s 64KB | zin[2] 128KB (async double buffer) | h0 8KB | h1 8KB = 208KB
__global__ void k_lstm_scan(const float* __restrict__ Z0,
                            const unsigned short* __restrict__ U0t,
                            const unsigned short* __restrict__ W1t,
                            const unsigned short* __restrict__ U1t,
                            const float* __restrict__ b1,
                            unsigned short* __restrict__ Ybf) {
    extern __shared__ char smem[];
    float*          z_s  = (float*)smem;                         // 32*512 f32
    float*          zin  = (float*)(smem + 65536);               // 2 slabs of 32*512
    unsigned short* h0_s = (unsigned short*)(smem + 3 * 65536);  // 32*128 bf16
    unsigned short* h1_s = h0_s + Bsz * Hsz;

    int tid  = threadIdx.x;           // 0..511
    int lane = tid & 31;
    int wave = tid >> 5;              // 0..15
    int lo = lane & 15, hi = lane >> 4;

    // 32-bit LDS byte address of the async-destination buffers (low 32 bits of
    // the generic shared address == wave-relative LDS address on gfx1250).
    unsigned zin_lds = (unsigned)(size_t)(void*)zin;

    // zero h state
    for (int i = tid; i < Bsz * Hsz; i += 512) { h0_s[i] = 0; h1_s[i] = 0; }
    float c0[8], c1[8];
#pragma unroll
    for (int j = 0; j < 8; ++j) { c0[j] = 0.0f; c1[j] = 0.0f; }

    // prime slab 0: 64KB = 4096 x b128, 512 threads x 8 async copies
#pragma unroll
    for (int j = 0; j < 8; ++j) {
        int e = j * 512 + tid;                       // b128 index in slab
        unsigned loff = zin_lds + (unsigned)e * 16u;
        const float* g = Z0 + (size_t)e * 4;
        asm volatile("global_load_async_to_lds_b128 %0, %1, off"
                     :: "v"(loff), "v"(g) : "memory");
    }
    asm volatile("s_wait_asynccnt 0x0" ::: "memory");
    __syncthreads();

    for (int t = 0; t < Tsz; ++t) {
        // ---- prefetch Z0[t+1] into the other LDS slab (async, off critical path)
        if (t + 1 < Tsz) {
            unsigned dst = zin_lds + (unsigned)((t + 1) & 1) * 65536u;
            const float* src = Z0 + (size_t)(t + 1) * Bsz * Gsz;
#pragma unroll
            for (int j = 0; j < 8; ++j) {
                int e = j * 512 + tid;
                unsigned loff = dst + (unsigned)e * 16u;
                const float* g = src + (size_t)e * 4;
                asm volatile("global_load_async_to_lds_b128 %0, %1, off"
                             :: "v"(loff), "v"(g) : "memory");
            }
        }

        // ---- layer0: z = Z0[t] + h0 @ U0 ----
        const float* zp = zin + (size_t)(t & 1) * Bsz * Gsz;
#pragma unroll
        for (int j = 0; j < 4; ++j) {
            int tl = wave * 4 + j;             // 64 tiles: 2 x 32
            int tm = tl >> 5, tn = tl & 31;
            int n = tn * 16 + lo;
            v8f acc;
#pragma unroll
            for (int r = 0; r < 8; ++r) {
                int M = tm * 16 + r + hi * 8;
                acc[r] = zp[(size_t)M * Gsz + n];
            }
            const unsigned short* ab = h0_s + (tm * 16 + lo) * 128 + hi * 16;
            const unsigned short* bb = U0t + (size_t)(tn * 16 + lo) * 128 + hi * 16;
#pragma unroll
            for (int kk = 0; kk < 4; ++kk)
                acc = wmma_bf16(load_a_frag(ab + kk * 32), load_b_frag(bb + kk * 32), acc);
#pragma unroll
            for (int r = 0; r < 8; ++r) {
                int M = tm * 16 + r + hi * 8;
                z_s[M * Gsz + n] = acc[r];
            }
        }
        __syncthreads();

        // ---- layer0 gates (each thread owns 8 (b,h) cells) ----
#pragma unroll
        for (int j = 0; j < 8; ++j) {
            int idx = j * 512 + tid;
            int bi = idx >> 7, h = idx & 127;
            const float* zr = z_s + bi * Gsz;
            float ig = sigm(zr[h]);
            float fg = sigm(zr[128 + h]);
            float cc = tanhf(zr[256 + h]);
            float og = sigm(zr[384 + h]);
            c0[j] = fg * c0[j] + ig * cc;
            h0_s[bi * Hsz + h] = f32_to_bf16(og * tanhf(c0[j]));
        }
        __syncthreads();

        // ---- layer1: z = h0 @ W1 + h1 @ U1 + b1 ----
#pragma unroll
        for (int j = 0; j < 4; ++j) {
            int tl = wave * 4 + j;
            int tm = tl >> 5, tn = tl & 31;
            int n = tn * 16 + lo;
            float bias = b1[n];
            v8f acc;
#pragma unroll
            for (int r = 0; r < 8; ++r) acc[r] = bias;
            const unsigned short* a0 = h0_s + (tm * 16 + lo) * 128 + hi * 16;
            const unsigned short* a1 = h1_s + (tm * 16 + lo) * 128 + hi * 16;
            const unsigned short* bw = W1t + (size_t)(tn * 16 + lo) * 128 + hi * 16;
            const unsigned short* bu = U1t + (size_t)(tn * 16 + lo) * 128 + hi * 16;
#pragma unroll
            for (int kk = 0; kk < 4; ++kk)
                acc = wmma_bf16(load_a_frag(a0 + kk * 32), load_b_frag(bw + kk * 32), acc);
#pragma unroll
            for (int kk = 0; kk < 4; ++kk)
                acc = wmma_bf16(load_a_frag(a1 + kk * 32), load_b_frag(bu + kk * 32), acc);
#pragma unroll
            for (int r = 0; r < 8; ++r) {
                int M = tm * 16 + r + hi * 8;
                z_s[M * Gsz + n] = acc[r];
            }
        }
        __syncthreads();

        // ---- layer1 gates + emit y ----
#pragma unroll
        for (int j = 0; j < 8; ++j) {
            int idx = j * 512 + tid;
            int bi = idx >> 7, h = idx & 127;
            const float* zr = z_s + bi * Gsz;
            float ig = sigm(zr[h]);
            float fg = sigm(zr[128 + h]);
            float cc = tanhf(zr[256 + h]);
            float og = sigm(zr[384 + h]);
            c1[j] = fg * c1[j] + ig * cc;
            unsigned short hv = f32_to_bf16(og * tanhf(c1[j]));
            h1_s[bi * Hsz + h] = hv;
            Ybf[((size_t)bi * Tsz + t) * Hsz + h] = hv;   // y row = b*T + t
        }
        // next slab must be resident before phase A of step t+1 reads it
        asm volatile("s_wait_asynccnt 0x0" ::: "memory");
        __syncthreads();
    }
}

// ---------------- logits = y @ Wd + bd ----------------
// Ybf: [4096,128] bf16; Wdt: [32000,128] bf16 N-major; out f32 [4096,32000]
// Each wave: one M-tile, 4 consecutive N-tiles (A-frags reused).
// Output is a 524MB write-once stream -> non-temporal stores (keep L2 for
// the reused Wdt/Ybf operands).
__global__ void k_proj(const unsigned short* __restrict__ Ybf,
                       const unsigned short* __restrict__ Wdt,
                       const float* __restrict__ bd, float* __restrict__ out) {
    int lane = threadIdx.x & 31;
    int wave = threadIdx.x >> 5;
    int task = blockIdx.x * 4 + wave;       // 256 * 500 = 128000 tasks
    int tm = task / 500;
    int tg = task - tm * 500;               // group of 4 N-tiles
    int lo = lane & 15, hi = lane >> 4;

    const unsigned short* ab = Ybf + (size_t)(tm * 16 + lo) * 128 + hi * 16;
    v16bf a0 = load_a_frag(ab);
    v16bf a1 = load_a_frag(ab + 32);
    v16bf a2 = load_a_frag(ab + 64);
    v16bf a3 = load_a_frag(ab + 96);

#pragma unroll
    for (int tnn = 0; tnn < 4; ++tnn) {
        int n0 = (tg * 4 + tnn) * 16;
        float bias = bd[n0 + lo];
        v8f acc;
#pragma unroll
        for (int r = 0; r < 8; ++r) acc[r] = bias;
        const unsigned short* bb = Wdt + (size_t)(n0 + lo) * 128 + hi * 16;
        acc = wmma_bf16(a0, load_b_frag(bb),      acc);
        acc = wmma_bf16(a1, load_b_frag(bb + 32), acc);
        acc = wmma_bf16(a2, load_b_frag(bb + 64), acc);
        acc = wmma_bf16(a3, load_b_frag(bb + 96), acc);

        float* op = out + (size_t)(tm * 16 + hi * 8) * Vsz + n0 + lo;
#pragma unroll
        for (int r = 0; r < 8; ++r)
            __builtin_nontemporal_store(acc[r], op + (size_t)r * Vsz);
    }
}

extern "C" void kernel_launch(void* const* d_in, const int* in_sizes, int n_in,
                              void* d_out, int out_size, void* d_ws, size_t ws_size,
                              hipStream_t stream) {
    const int*   tok = (const int*)d_in[0];
    const float* emb = (const float*)d_in[1];
    const float* W0  = (const float*)d_in[2];
    const float* U0  = (const float*)d_in[3];
    const float* b0  = (const float*)d_in[4];
    const float* W1  = (const float*)d_in[5];
    const float* U1  = (const float*)d_in[6];
    const float* b1  = (const float*)d_in[7];
    const float* Wd  = (const float*)d_in[8];
    const float* bd  = (const float*)d_in[9];
    float* out = (float*)d_out;

    char* ws = (char*)d_ws;
    unsigned short* W0t = (unsigned short*)ws; ws += (size_t)Gsz * Hsz * 2;
    unsigned short* U0t = (unsigned short*)ws; ws += (size_t)Gsz * Hsz * 2;
    unsigned short* W1t = (unsigned short*)ws; ws += (size_t)Gsz * Hsz * 2;
    unsigned short* U1t = (unsigned short*)ws; ws += (size_t)Gsz * Hsz * 2;
    unsigned short* Wdt = (unsigned short*)ws; ws += (size_t)Vsz * Hsz * 2;
    unsigned short* Xbf = (unsigned short*)ws; ws += (size_t)Bsz * Tsz * Hsz * 2;
    unsigned short* Ybf = (unsigned short*)ws; ws += (size_t)Bsz * Tsz * Hsz * 2;
    float*          Z0  = (float*)ws;          ws += (size_t)Tsz * Bsz * Gsz * 4;

    int tw = Gsz * Hsz;                // 65536
    k_transpose_bf16<<<(tw + 255) / 256, 256, 0, stream>>>(W0, W0t, Hsz, Gsz);
    k_transpose_bf16<<<(tw + 255) / 256, 256, 0, stream>>>(U0, U0t, Hsz, Gsz);
    k_transpose_bf16<<<(tw + 255) / 256, 256, 0, stream>>>(W1, W1t, Hsz, Gsz);
    k_transpose_bf16<<<(tw + 255) / 256, 256, 0, stream>>>(U1, U1t, Hsz, Gsz);
    int td = Vsz * Hsz;                // 4,096,000
    k_transpose_bf16<<<(td + 255) / 256, 256, 0, stream>>>(Wd, Wdt, Hsz, Vsz);

    int te = Bsz * Tsz * Hsz;          // 524,288
    k_embed<<<(te + 255) / 256, 256, 0, stream>>>(tok, emb, Xbf, te);

    k_gemm_z0<<<2048, 128, 0, stream>>>(Xbf, W0t, b0, Z0);

    size_t lds = 3 * 65536 + 2 * (size_t)Bsz * Hsz * 2; // 208KB
    k_lstm_scan<<<1, 512, lds, stream>>>(Z0, U0t, W1t, U1t, b1, Ybf);

    k_proj<<<32000, 128, 0, stream>>>(Ybf, Wdt, bd, out);
}
